// Transformer_Encoder_Layer_25580825215162
// MI455X (gfx1250) — compile-verified, matrix operand order checked by reading, not run on hardware
//
#include <hip/hip_runtime.h>

#define Hh   160
#define NHh  8
#define HDd  20
#define Ss   256
#define Bb   4
#define FFf  640
#define TBL  1025
#define MAXL 512

#define TM 32
#define TN 64
#define TK 32

typedef __attribute__((ext_vector_type(16))) __bf16 v16bf;
typedef __attribute__((ext_vector_type(8)))  __bf16 v8bf;
typedef __attribute__((ext_vector_type(4)))  __bf16 v4bf;
typedef __attribute__((ext_vector_type(8)))  float  v8f;
typedef __attribute__((ext_vector_type(4)))  unsigned int v4u;
typedef __attribute__((ext_vector_type(8)))  int v8i;
typedef __attribute__((ext_vector_type(4)))  int v4i;

__device__ inline float wave_sum(float v) {
#pragma unroll
  for (int o = 16; o > 0; o >>= 1) v += __shfl_xor(v, o, 32);
  return v;
}
__device__ inline float wave_max(float v) {
#pragma unroll
  for (int o = 16; o > 0; o >>= 1) { float t = __shfl_xor(v, o, 32); v = v > t ? v : t; }
  return v;
}

// ---------------------------------------------------------------------------
// TDM: async-copy a 2-D bf16 tile (rows x 32 elems, row stride ld_elems) from
// global memory into LDS at lds_off. Descriptor per CDNA5 ISA ch.8 (D#):
// group0 = {count=1, lds_addr, global_addr, type=2}; group1 = {data_size=2B,
// tensor_dim = tile_dim (no OOB clipping; interior tiles only), stride}.
// 6-arg builtin variant (clang-23 / therock headers).
// ---------------------------------------------------------------------------
__device__ inline void tdm_load_2d_bf16(const __bf16* g, unsigned lds_off,
                                        int rows, int ld_elems)
{
  unsigned long long ga = (unsigned long long)(uintptr_t)g;
  v4u g0;
  g0[0] = 1u;                                              // count=1, user mode
  g0[1] = lds_off;                                         // lds_addr
  g0[2] = (unsigned)ga;                                    // global_addr lo
  g0[3] = (unsigned)((ga >> 32) & 0x01FFFFFFu) | 0x80000000u; // addr hi | type=2
  v8i g1;
  g1[0] = 1 << 16;                        // workgroup_mask=0, data_size=1 (2B)
  g1[1] = (int)(32u << 16);               // tensor_dim0 = 32
  g1[2] = (int)((unsigned)rows << 16);    // tensor_dim1 = rows
  g1[3] = (int)(32u << 16);               // tile_dim0 = 32
  g1[4] = rows;                           // tile_dim1 = rows, tile_dim2 = 0
  g1[5] = ld_elems;                       // tensor_dim0_stride (48b, low 32)
  g1[6] = 0;
  g1[7] = 0;
  v4i z4 = {0, 0, 0, 0};
  v8i z8 = {0, 0, 0, 0, 0, 0, 0, 0};
  __builtin_amdgcn_tensor_load_to_lds(g0, g1, z4, z4, z8, 0);
}

// ---------------------------------------------------------------------------
// Batched GEMM, all-bf16 operands: A is [M][K] (lda), B is pre-transposed
// [N][K] (ldb). 256-thread block (8 waves), macro tile 32x64x32; each wave
// computes one 16x16 WMMA tile, fragments fetched as 2x ds_load_b128.
// Interior tiles staged by the Tensor Data Mover (wave 0 issues, TENSORcnt
// wait); edge tiles staged by guarded VGPR loads.
// C(f32) = alpha*(A@B) + (accum ? C : 0) + bias, optional ReLU; optional bf16
// shadow copy Cbf for the next GEMM in the chain.
// ---------------------------------------------------------------------------
__global__ __launch_bounds__(256)
void gemm_wmma_bf16(const __bf16* __restrict__ A, long sA1, long sA2, int lda,
                    const __bf16* __restrict__ Bt, long sB1, long sB2, int ldb,
                    const float* __restrict__ bias,
                    float* __restrict__ C, __bf16* __restrict__ Cbf,
                    long sC1, long sC2, int ldc,
                    int M, int N, int K, int nb2,
                    float alpha, int accum, int do_relu)
{
  __shared__ alignas(16) __bf16 Ash[TM * TK];
  __shared__ alignas(16) __bf16 Bsh[TN * TK];

  const int tid  = threadIdx.x;
  const int lane = tid & 31;
  const int wid  = tid >> 5;      // 0..7
  const int wm   = wid & 1;       // 2 wave-tiles along M
  const int wn   = wid >> 1;      // 4 wave-tiles along N
  const int half = lane >> 4;
  const int mn   = lane & 15;

  const int n0 = blockIdx.x * TN;
  const int m0 = blockIdx.y * TM;
  const int z  = blockIdx.z;
  const int z1 = z / nb2, z2 = z % nb2;
  A  += z1 * sA1 + z2 * sA2;
  Bt += z1 * sB1 + z2 * sB2;
  C  += z1 * sC1 + z2 * sC2;
  if (Cbf) Cbf += z1 * sC1 + z2 * sC2;

  const unsigned ldsA = (unsigned)(uintptr_t)&Ash[0];
  const unsigned ldsB = (unsigned)(uintptr_t)&Bsh[0];

  const int am = tid >> 3;            // edge A fill: row 0..31
  const int ak = (tid & 7) * 4;       //              k4 chunk
  const int bn = tid >> 2;            // edge B fill: n 0..63
  const int bk = (tid & 3) * 8;       //              k8 chunk

  v8f acc = {};
  for (int kk = 0; kk < K; kk += TK) {
    __syncthreads();
    const bool intA = (m0 + TM <= M) && (kk + TK <= K);
    const bool intB = (n0 + TN <= N) && (kk + TK <= K);
    if (wid == 0) {
      if (intA) tdm_load_2d_bf16(A + (long)m0 * lda + kk, ldsA, TM, lda);
      if (intB) tdm_load_2d_bf16(Bt + (long)n0 * ldb + kk, ldsB, TN, ldb);
    }
    if (!intA) {
      const int gm = m0 + am, gk = kk + ak;
      v4bf pk;
#pragma unroll
      for (int i = 0; i < 4; ++i)
        pk[i] = (gm < M && gk + i < K) ? A[(long)gm * lda + gk + i] : (__bf16)0.f;
      *(v4bf*)&Ash[am * TK + ak] = pk;
    }
    if (!intB) {
      const int gn = n0 + bn, gk = kk + bk;
      v8bf pk;
#pragma unroll
      for (int i = 0; i < 8; ++i)
        pk[i] = (gn < N && gk + i < K) ? Bt[(long)gn * ldb + gk + i] : (__bf16)0.f;
      *(v8bf*)&Bsh[bn * TK + bk] = pk;
    }
    if (wid == 0) __builtin_amdgcn_s_wait_tensorcnt(0);
    __syncthreads();
    // ---- fragment fetch (2x ds_load_b128 per operand) + WMMA ----
    const v8bf* arp = (const v8bf*)&Ash[(wm * 16 + mn) * TK];
    const v8bf* brp = (const v8bf*)&Bsh[(wn * 16 + mn) * TK];
    const v8bf alo = arp[half], ahi = arp[half + 2];
    const v8bf blo = brp[half], bhi = brp[half + 2];
    v16bf av, bvv;
#pragma unroll
    for (int e = 0; e < 8; ++e) {
      av[e]  = alo[e]; av[e + 8]  = ahi[e];
      bvv[e] = blo[e]; bvv[e + 8] = bhi[e];
    }
    acc = __builtin_amdgcn_wmma_f32_16x16x32_bf16(false, av, false, bvv,
                                                  (short)0, acc, false, false);
  }

  const int gn = n0 + wn * 16 + mn;
  const float bs = (bias && gn < N) ? bias[gn] : 0.f;
#pragma unroll
  for (int r = 0; r < 8; ++r) {
    const int gm = m0 + wm * 16 + r + half * 8;
    if (gm < M && gn < N) {
      float* cp  = &C[(long)gm * ldc + gn];
      float  res = alpha * acc[r] + bs + (accum ? *cp : 0.f);
      if (do_relu && res < 0.f) res = 0.f;
      *cp = res;
      if (Cbf) Cbf[(long)gm * ldc + gn] = (__bf16)res;
    }
  }
}

// flat f32 -> bf16
__global__ void cvt_bf16(const float* __restrict__ in, __bf16* __restrict__ out, int n)
{
  int idx = blockIdx.x * blockDim.x + threadIdx.x;
  if (idx < n) out[idx] = (__bf16)in[idx];
}

// src f32 [K,N] -> dst bf16 [N,K]
__global__ void transpose_cvt(const float* __restrict__ src, __bf16* __restrict__ dst,
                              int K, int N)
{
  int idx = blockIdx.x * blockDim.x + threadIdx.x;
  if (idx < K * N) {
    const int n = idx / K, k = idx % K;
    dst[idx] = (__bf16)src[(long)k * N + n];
  }
}

// xvT[b,h,d,j] = xv[b,j,h*HD+d]   (bf16)
__global__ void xv_transpose(const float* __restrict__ xv, __bf16* __restrict__ xvT)
{
  int idx = blockIdx.x * blockDim.x + threadIdx.x;
  const int total = Bb * NHh * HDd * Ss;
  if (idx < total) {
    const int j = idx % Ss;
    const int d = (idx / Ss) % HDd;
    const int h = (idx / (Ss * HDd)) % NHh;
    const int b = idx / (Ss * HDd * NHh);
    xvT[idx] = (__bf16)xv[((long)(b * Ss + j)) * Hh + h * HDd + d];
  }
}

// out[r,c] = in[r,c] + vec[c]; optional f32 and bf16 outputs
__global__ void add_rowvec(const float* __restrict__ in, const float* __restrict__ vec,
                           float* __restrict__ out, __bf16* __restrict__ out_bf,
                           int total, int cols)
{
  int idx = blockIdx.x * blockDim.x + threadIdx.x;
  if (idx < total) {
    const float r = in[idx] + vec[idx % cols];
    if (out)    out[idx]    = r;
    if (out_bf) out_bf[idx] = (__bf16)r;
  }
}

// t[b,i,h,e] = sum_d Wr[e, h*HD+d] * xqv[b,i,h*HD+d];  cbr[b,i,h] = (q+v).br_h
__global__ __launch_bounds__(32)
void t_kernel(const float* __restrict__ xqv, const float* __restrict__ Wr,
              const float* __restrict__ br, float* __restrict__ tbuf,
              float* __restrict__ cbr)
{
  const int bih = blockIdx.x;            // (b*S+i)*NH + h
  const int h   = bih % NHh;
  const int bi  = bih / NHh;
  const int lane = threadIdx.x;
  const float* qv = xqv + (long)bi * Hh + h * HDd;
  float q[HDd];
#pragma unroll
  for (int d = 0; d < HDd; ++d) q[d] = qv[d];
#pragma unroll
  for (int r = 0; r < 5; ++r) {
    const int e = lane + 32 * r;
    const float* w = Wr + (long)e * Hh + h * HDd;
    float s = 0.f;
#pragma unroll
    for (int d = 0; d < HDd; ++d) s += w[d] * q[d];
    tbuf[(long)bih * Hh + e] = s;
  }
  float p = (lane < HDd) ? qv[lane] * br[h * HDd + lane] : 0.f;
  p = wave_sum(p);
  if (lane == 0) cbr[bih] = p;
}

// B_D term: materialize rel[b,i,j,:] in registers from folded P tables, dot
// with t[b,i,h,:] from LDS, write scaled scores sc[b,h,i,j].
__global__ __launch_bounds__(256)
void bd_kernel(const float* __restrict__ P, const float* __restrict__ b_fus,
               const int* __restrict__ pos_s, const int* __restrict__ pos_e,
               const float* __restrict__ tbuf, const float* __restrict__ cbr,
               float* __restrict__ sc, float inv_sqrt_hd)
{
  __shared__ float t_sh[NHh * Hh];
  __shared__ float cbr_sh[NHh];
  __shared__ float bf_sh[Hh];
  __shared__ int   ps_sh[Ss], pe_sh[Ss];
  const int bi = blockIdx.x;             // b*S + i
  const int b  = bi / Ss, i = bi % Ss;
  const int tid = threadIdx.x;
#pragma unroll
  for (int r = 0; r < (NHh * Hh) / 256; ++r)
    t_sh[tid + r * 256] = tbuf[(long)bi * NHh * Hh + tid + r * 256];
  if (tid < NHh) cbr_sh[tid] = cbr[bi * NHh + tid];
  if (tid < Hh)  bf_sh[tid]  = b_fus[tid];
  ps_sh[tid] = pos_s[b * Ss + tid];
  pe_sh[tid] = pos_e[b * Ss + tid];
  __syncthreads();

  const int lane = tid & 31, w = tid >> 5;
  const int psi = ps_sh[i], pei = pe_sh[i];
  const float* P0 = P;
  const float* P1 = P + 1L * TBL * Hh;
  const float* P2 = P + 2L * TBL * Hh;
  const float* P3 = P + 3L * TBL * Hh;
  for (int j = w; j < Ss; j += 8) {
    const int psj = ps_sh[j], pej = pe_sh[j];
    const long dss = (long)(psi - psj + MAXL) * Hh;
    const long dse = (long)(psi - pej + MAXL) * Hh;
    const long des = (long)(pei - psj + MAXL) * Hh;
    const long dee = (long)(pei - pej + MAXL) * Hh;
    if (j + 8 < Ss) {  // prefetch next iteration's table rows into cache
      const int jn = j + 8;
      const int psn = ps_sh[jn], pen = pe_sh[jn];
      __builtin_prefetch(&P0[(long)(psi - psn + MAXL) * Hh + lane], 0, 1);
      __builtin_prefetch(&P1[(long)(psi - pen + MAXL) * Hh + lane], 0, 1);
      __builtin_prefetch(&P2[(long)(pei - psn + MAXL) * Hh + lane], 0, 1);
      __builtin_prefetch(&P3[(long)(pei - pen + MAXL) * Hh + lane], 0, 1);
    }
    float rel[5];
#pragma unroll
    for (int r = 0; r < 5; ++r) {
      const int e = lane + 32 * r;
      float s = P0[dss + e] + P1[dse + e] + P2[des + e] + P3[dee + e] + bf_sh[e];
      rel[r] = s > 0.f ? s : 0.f;
    }
#pragma unroll
    for (int h = 0; h < NHh; ++h) {
      float p = 0.f;
#pragma unroll
      for (int r = 0; r < 5; ++r) p += rel[r] * t_sh[h * Hh + lane + 32 * r];
      p = wave_sum(p);
      if (lane == 0)
        sc[(((long)b * NHh + h) * Ss + i) * Ss + j] = (p + cbr_sh[h]) * inv_sqrt_hd;
    }
  }
}

// masked softmax over last dim (S=256), one wave per row; emits bf16 probs.
__global__ __launch_bounds__(32)
void softmax_kernel(const float* __restrict__ sc, __bf16* __restrict__ attn_bf,
                    const int* __restrict__ seq_len, const int* __restrict__ lex_num)
{
  const long row = blockIdx.x;           // B*NH*S rows
  const int  b   = (int)(row / (NHh * Ss));
  const int  lim = seq_len[b] + lex_num[0];
  const float* p = sc + row * Ss;
  __bf16* q = attn_bf + row * Ss;
  const int lane = threadIdx.x;
  float v[8];
  float m = -3.0e38f;
#pragma unroll
  for (int r = 0; r < 8; ++r) {
    const int j = lane + 32 * r;
    v[r] = (j < lim) ? p[j] : -3.0e38f;
    m = fmaxf(m, v[r]);
  }
  m = wave_max(m);
  float s = 0.f;
#pragma unroll
  for (int r = 0; r < 8; ++r) {
    const int j = lane + 32 * r;
    v[r] = (j < lim) ? __expf(v[r] - m) : 0.f;
    s += v[r];
  }
  s = wave_sum(s);
  const float inv = 1.f / s;
#pragma unroll
  for (int r = 0; r < 8; ++r) q[lane + 32 * r] = (__bf16)(v[r] * inv);
}

// y = LN(pre_scale * x) * g + b over rows of H=160; optional bf16 shadow.
__global__ __launch_bounds__(32)
void ln_kernel(const float* __restrict__ in, const float* __restrict__ g,
               const float* __restrict__ bvec, float* __restrict__ out,
               __bf16* __restrict__ out_bf, float pre_scale)
{
  const long row = blockIdx.x;
  const float* x = in + row * Hh;
  const int lane = threadIdx.x;
  float v[5];
  float s = 0.f;
#pragma unroll
  for (int r = 0; r < 5; ++r) { v[r] = pre_scale * x[lane + 32 * r]; s += v[r]; }
  s = wave_sum(s);
  const float mu = s * (1.0f / Hh);
  float vs = 0.f;
#pragma unroll
  for (int r = 0; r < 5; ++r) { const float d = v[r] - mu; vs += d * d; }
  vs = wave_sum(vs);
  const float rs = rsqrtf(vs * (1.0f / Hh) + 1e-5f);
#pragma unroll
  for (int r = 0; r < 5; ++r) {
    const int e = lane + 32 * r;
    const float y = (v[r] - mu) * rs * g[e] + bvec[e];
    out[row * Hh + e] = y;
    if (out_bf) out_bf[row * Hh + e] = (__bf16)y;
  }
}

extern "C" void kernel_launch(void* const* d_in, const int* in_sizes, int n_in,
                              void* d_out, int out_size, void* d_ws, size_t ws_size,
                              hipStream_t stream)
{
  const float* inp    = (const float*)d_in[0];
  const int*   pos_s  = (const int*)d_in[1];
  const int*   pos_e  = (const int*)d_in[2];
  const int*   seqlen = (const int*)d_in[3];
  const int*   lexnum = (const int*)d_in[4];
  const float* pe_t[4] = { (const float*)d_in[5], (const float*)d_in[6],
                           (const float*)d_in[7], (const float*)d_in[8] };
  const float* W_fus = (const float*)d_in[9];
  const float* b_fus = (const float*)d_in[10];
  const float* Wq = (const float*)d_in[11]; const float* bq = (const float*)d_in[12];
  const float* Wk = (const float*)d_in[13]; const float* bk = (const float*)d_in[14];
  const float* Wv = (const float*)d_in[15]; const float* bv = (const float*)d_in[16];
  const float* Wr = (const float*)d_in[17]; const float* br = (const float*)d_in[18];
  const float* uvec = (const float*)d_in[19];
  const float* vvec = (const float*)d_in[20];
  const float* W_fin = (const float*)d_in[21]; const float* b_fin = (const float*)d_in[22];
  const float* ln1_g = (const float*)d_in[23]; const float* ln1_b = (const float*)d_in[24];
  const float* W1 = (const float*)d_in[25]; const float* b1 = (const float*)d_in[26];
  const float* W2 = (const float*)d_in[27]; const float* b2 = (const float*)d_in[28];
  const float* ln2_g = (const float*)d_in[29]; const float* ln2_b = (const float*)d_in[30];

  const int BS  = Bb * Ss;                // 1024 rows
  const int HH  = Hh * Hh;                // 25600

  // ---- f32 workspace ----
  float* ws = (float*)d_ws;
  long off = 0;
  float* P       = ws + off;  off += 4L * TBL * Hh;
  float* xq      = ws + off;  off += (long)BS * Hh;
  float* xk      = ws + off;  off += (long)BS * Hh;
  float* xv      = ws + off;  off += (long)BS * Hh;
  float* xqv     = ws + off;  off += (long)BS * Hh;
  float* tbuf    = ws + off;  off += (long)BS * NHh * Hh;
  float* cbr     = ws + off;  off += (long)BS * NHh;
  float* sc      = ws + off;  off += (long)Bb * NHh * Ss * Ss;
  float* attnout = ws + off;  off += (long)BS * Hh;
  float* ao      = ws + off;  off += (long)BS * Hh;
  float* x1      = ws + off;  off += (long)BS * Hh;
  float* f1      = ws + off;  off += (long)BS * FFf;
  float* f2      = ws + off;  off += (long)BS * Hh;

  // ---- bf16 workspace (carved after f32 region) ----
  __bf16* wb = (__bf16*)(ws + off);
  long ob = 0;
  __bf16* inp_bf   = wb + ob;  ob += (long)BS * Hh;
  __bf16* pe_bf    = wb + ob;  ob += 4L * TBL * Hh;
  __bf16* WfusT    = wb + ob;  ob += 4L * HH;
  __bf16* WqT      = wb + ob;  ob += HH;
  __bf16* WkT      = wb + ob;  ob += HH;
  __bf16* WvT      = wb + ob;  ob += HH;
  __bf16* WfinT    = wb + ob;  ob += HH;
  __bf16* W1T      = wb + ob;  ob += (long)Hh * FFf;
  __bf16* W2T      = wb + ob;  ob += (long)Hh * FFf;
  __bf16* xqu_bf   = wb + ob;  ob += (long)BS * Hh;
  __bf16* xk_bf    = wb + ob;  ob += (long)BS * Hh;
  __bf16* xvT_bf   = wb + ob;  ob += (long)BS * Hh;
  __bf16* attn_bf  = wb + ob;  ob += (long)Bb * NHh * Ss * Ss;
  __bf16* aout_bf  = wb + ob;  ob += (long)BS * Hh;
  __bf16* x1_bf    = wb + ob;  ob += (long)BS * Hh;
  __bf16* f1_bf    = wb + ob;  ob += (long)BS * FFf;

  const float invsq = 0.22360679774997896f;  // 1/sqrt(HD)

  auto gemm = [&](const __bf16* A, long sA1, long sA2, int lda,
                  const __bf16* Bm, long sB1, long sB2, int ldb,
                  const float* bias, float* C, __bf16* Cbf,
                  long sC1, long sC2, int ldc,
                  int M, int N, int K, int batches, int nb2,
                  float alpha, int accum, int relu) {
    dim3 grid((N + TN - 1) / TN, (M + TM - 1) / TM, batches);
    gemm_wmma_bf16<<<grid, 256, 0, stream>>>(A, sA1, sA2, lda,
                                             Bm, sB1, sB2, ldb, bias,
                                             C, Cbf, sC1, sC2, ldc, M, N, K, nb2,
                                             alpha, accum, relu);
  };
  auto cvt = [&](const float* src, __bf16* dst, long n) {
    cvt_bf16<<<(int)((n + 255) / 256), 256, 0, stream>>>(src, dst, (int)n);
  };
  auto tpose = [&](const float* src, __bf16* dst, int K, int N) {
    transpose_cvt<<<(K * N + 255) / 256, 256, 0, stream>>>(src, dst, K, N);
  };

  // 0) one-time operand conversion: activations flat, weights transposed [N][K]
  cvt(inp, inp_bf, (long)BS * Hh);
  for (int t = 0; t < 4; ++t) cvt(pe_t[t], pe_bf + (long)t * TBL * Hh, (long)TBL * Hh);
  for (int t = 0; t < 4; ++t) tpose(W_fus + (long)t * HH, WfusT + (long)t * HH, Hh, Hh);
  tpose(Wq, WqT, Hh, Hh);   tpose(Wk, WkT, Hh, Hh);
  tpose(Wv, WvT, Hh, Hh);   tpose(W_fin, WfinT, Hh, Hh);
  tpose(W1, W1T, Hh, FFf);  // [160,640] -> [640,160]
  tpose(W2, W2T, FFf, Hh);  // [640,160] -> [160,640]

  // 1) Fold W_fus into the four pe tables: P_t = pe_t @ W_fus[t*H:(t+1)*H, :]
  for (int t = 0; t < 4; ++t)
    gemm(pe_bf + (long)t * TBL * Hh, 0, 0, Hh, WfusT + (long)t * HH, 0, 0, Hh,
         nullptr, P + (long)t * TBL * Hh, nullptr, 0, 0, Hh,
         TBL, Hh, Hh, 1, 1, 1.f, 0, 0);

  // 2) Q/K/V projections (K emits bf16 shadow for A_C; V stays f32)
  gemm(inp_bf, 0, 0, Hh, WqT, 0, 0, Hh, bq, xq, nullptr, 0, 0, Hh, BS, Hh, Hh, 1, 1, 1.f, 0, 0);
  gemm(inp_bf, 0, 0, Hh, WkT, 0, 0, Hh, bk, xk, xk_bf,  0, 0, Hh, BS, Hh, Hh, 1, 1, 1.f, 0, 0);
  gemm(inp_bf, 0, 0, Hh, WvT, 0, 0, Hh, bv, xv, nullptr, 0, 0, Hh, BS, Hh, Hh, 1, 1, 1.f, 0, 0);

  // 3) q+u (bf16, A of A_C) and q+v (f32, input to t_kernel)
  add_rowvec<<<(BS * Hh + 255) / 256, 256, 0, stream>>>(xq, uvec, nullptr, xqu_bf, BS * Hh, Hh);
  add_rowvec<<<(BS * Hh + 255) / 256, 256, 0, stream>>>(xq, vvec, xqv, nullptr, BS * Hh, Hh);

  // 3b) transpose V for attn@V: xvT[b,h,d,j]
  xv_transpose<<<(BS * Hh + 255) / 256, 256, 0, stream>>>(xv, xvT_bf);

  // 4) Fold Wr/br into (q+v): t[b,i,h,:] and constant term
  t_kernel<<<BS * NHh, 32, 0, stream>>>(xqv, Wr, br, tbuf, cbr);

  // 5) B_D via gathered folded tables -> scaled scores
  bd_kernel<<<BS, 256, 0, stream>>>(P, b_fus, pos_s, pos_e, tbuf, cbr, sc, invsq);

  // 6) A_C accumulated on top: sc += invsq * (q+u) @ k^T  (batched over b,h)
  gemm(xqu_bf, (long)Ss * Hh, HDd, Hh,
       xk_bf,  (long)Ss * Hh, HDd, Hh,
       nullptr, sc, nullptr, (long)NHh * Ss * Ss, (long)Ss * Ss, Ss,
       Ss, Ss, HDd, Bb * NHh, NHh, invsq, 1, 0);

  // 7) masked softmax -> bf16 attention probs
  softmax_kernel<<<Bb * NHh * Ss, 32, 0, stream>>>(sc, attn_bf, seqlen, lexnum);

  // 8) attn @ V  (batched over b,h; writes head slice of attnout)
  gemm(attn_bf, (long)NHh * Ss * Ss, (long)Ss * Ss, Ss,
       xvT_bf,  (long)NHh * HDd * Ss, (long)HDd * Ss, Ss,
       nullptr, attnout, aout_bf, (long)Ss * Hh, HDd, Hh,
       Ss, HDd, Ss, Bb * NHh, NHh, 1.f, 0, 0);

  // 9) final projection
  gemm(aout_bf, 0, 0, Hh, WfinT, 0, 0, Hh, b_fin, ao, nullptr, 0, 0, Hh,
       BS, Hh, Hh, 1, 1, 1.f, 0, 0);

  // 10) LN(out+out)
  ln_kernel<<<BS, 32, 0, stream>>>(ao, ln1_g, ln1_b, x1, x1_bf, 2.0f);

  // 11) FFN
  gemm(x1_bf, 0, 0, Hh, W1T, 0, 0, Hh, b1, f1, f1_bf, 0, 0, FFf,
       BS, FFf, Hh, 1, 1, 1.f, 0, 1);
  gemm(f1_bf, 0, 0, FFf, W2T, 0, 0, FFf, b2, f2, nullptr, 0, 0, Hh,
       BS, Hh, FFf, 1, 1, 1.f, 0, 0);

  // 12) LN(ff+ff) -> output
  ln_kernel<<<BS, 32, 0, stream>>>(f2, ln2_g, ln2_b, (float*)d_out, nullptr, 2.0f);
}
